// AdditiveAttention_9036611191302
// MI455X (gfx1250) — matrix operand order repaired and checked by reading, not compiled, not run on hardware
//
#include <hip/hip_runtime.h>
#include <hip/hip_bf16.h>
#include <cmath>
#include <cstdint>

typedef __attribute__((ext_vector_type(16))) _Float16 v16h;
typedef __attribute__((ext_vector_type(8)))  _Float16 v8h;
typedef __attribute__((ext_vector_type(8)))  float    v8f;
typedef __attribute__((ext_vector_type(4)))  float    v4f;
typedef __attribute__((ext_vector_type(4)))  int      v4i;
typedef __attribute__((ext_vector_type(4)))  unsigned int u32x4;
typedef __attribute__((ext_vector_type(8)))  int      i32x8;
typedef __attribute__((ext_vector_type(4)))  int      i32x4;

#define B_  4
#define NQ  64
#define S_  1024
#define D_  512    // Q_DIM == K_DIM == HID == V dim
#define LDW 1024   // W1 row stride (Q_DIM + K_DIM)

#define TQ  4      // q rows per score block (kp reuse factor)
#define SCH 256    // s rows per score block
#define TR  8      // s rows per LDS tile
#define NT  (SCH / TR)

__device__ __forceinline__ float fast_tanh(float x) {
#if __has_builtin(__builtin_amdgcn_tanhf)
  return __builtin_amdgcn_tanhf(x);
#elif __has_builtin(__builtin_amdgcn_tanh_f32)
  return __builtin_amdgcn_tanh_f32(x);
#else
  return tanhf(x);
#endif
}

// ---- async global->LDS staging (gfx1250) ----------------------------------
#if __has_builtin(__builtin_amdgcn_global_load_async_to_lds_b128)
#define HAVE_ASYNC 1
#else
#define HAVE_ASYNC 0
#endif

// ---- Tensor Data Mover ----------------------------------------------------
#if __has_builtin(__builtin_amdgcn_tensor_load_to_lds) && \
    __has_builtin(__builtin_amdgcn_s_wait_tensorcnt)
#define HAVE_TDM 1
#else
#define HAVE_TDM 0
#endif

typedef __attribute__((address_space(1))) v4i as1_v4i;   // global (__device__)
typedef __attribute__((address_space(3))) v4i as3_v4i;   // LDS (__shared__)

__device__ __forceinline__ void g2l_b128(void* lds, const void* g) {
#if HAVE_ASYNC
  __builtin_amdgcn_global_load_async_to_lds_b128(
      (as1_v4i*)(uintptr_t)g, (as3_v4i*)(uintptr_t)lds, 0, 0);
#else
  *(v4f*)lds = *(const v4f*)g;   // fallback through VGPRs
#endif
}

__device__ __forceinline__ void wait_async() {
#if HAVE_ASYNC
#if __has_builtin(__builtin_amdgcn_s_wait_asynccnt)
  __builtin_amdgcn_s_wait_asynccnt(0);
#else
  asm volatile("s_wait_asynccnt 0" ::: "memory");
#endif
#endif
}

#if HAVE_TDM
// One flat 1-D TDM transfer: nelem f32 elements, global -> LDS.
// D# per CDNA5 ISA ch8: group0 = {count|flags, lds_addr, global_addr, type=2},
// group1 = {data_size=4B, tensor_dim0 = tile_dim0 = nelem, dim0_stride = nelem},
// groups 2/3 zero (<=2D tile).
__device__ __forceinline__ void tdm_load_1d(void* lds, const void* g,
                                            unsigned nelem) {
  const unsigned long long ga = (unsigned long long)(uintptr_t)g;
  const unsigned lds_off = (unsigned)(uintptr_t)lds;   // LDS byte offset
  u32x4 g0;
  g0[0] = 1u;                                          // count=1, user D#
  g0[1] = lds_off;                                     // D#.lds_addr
  g0[2] = (unsigned)ga;                                // global_addr[31:0]
  g0[3] = (unsigned)((ga >> 32) & 0x01FFFFFFu)         // global_addr[56:32]
        | (2u << 30);                                  // type = 2 ("image")
  i32x8 g1;
  g1[0] = (int)(2u << 16);                             // data_size = 4 bytes
  g1[1] = (int)((nelem & 0xFFFFu) << 16);              // tensor_dim0[15:0]  (bits 63:48)
  g1[2] = (int)((nelem >> 16) & 0xFFFFu);              // tensor_dim0[31:16] (bits 79:64)
  g1[3] = (int)((nelem & 0xFFFFu) << 16);              // tile_dim0 (bits 127:112)
  g1[4] = 0;                                           // tile_dim1/2 = 0 (unused)
  g1[5] = (int)nelem;                                  // tensor_dim0_stride[31:0]
  g1[6] = 0;
  g1[7] = 0;
  i32x4 g2 = {}, g3 = {};
#if __clang_major__ >= 23
  i32x8 g4 = {};
  __builtin_amdgcn_tensor_load_to_lds(g0, g1, g2, g3, g4, 0);
#else
  __builtin_amdgcn_tensor_load_to_lds(g0, g1, g2, g3, 0);
#endif
}
#endif

// Stage one TR x 512 f32 tile (contiguous 16 KB) into LDS.
__device__ __forceinline__ void stage_tile(float* ldsbuf, const float* src,
                                           int tid) {
#if HAVE_TDM
  if (tid < 32) tdm_load_1d(ldsbuf, src, TR * D_);     // wave 0 issues one TDM op
#else
#pragma unroll
  for (int j = 0; j < 4; ++j) {
    const int idx = j * 256 + tid;                     // float4 index in tile
    g2l_b128(ldsbuf + idx * 4, src + (size_t)idx * 4);
  }
#endif
}

__device__ __forceinline__ void wait_stage(int tid) {
#if HAVE_TDM
  if (tid < 32) __builtin_amdgcn_s_wait_tensorcnt(0);
#else
  wait_async();
#endif
}

// ---------------------------------------------------------------------------
// values (f32, [b][s][v]) -> vT (f16, [b][v][s])  -- LDS-tiled transpose
// ---------------------------------------------------------------------------
__global__ __launch_bounds__(256)
void transpose_cvt(const float* __restrict__ vals, _Float16* __restrict__ vT) {
  __shared__ _Float16 tile[32][33];
  const int b  = blockIdx.z;
  const int s0 = blockIdx.x * 32, v0 = blockIdx.y * 32;
  const float* src = vals + (size_t)b * S_ * D_;
  _Float16*    dst = vT   + (size_t)b * D_ * S_;
  const int tx = threadIdx.x, ty = threadIdx.y;  // 32 x 8
  for (int i = ty; i < 32; i += 8)
    tile[i][tx] = (_Float16)src[(size_t)(s0 + i) * D_ + (v0 + tx)];
  __syncthreads();
  for (int i = ty; i < 32; i += 8)
    dst[(size_t)(v0 + i) * S_ + (s0 + tx)] = tile[tx][i];
}

// ---------------------------------------------------------------------------
// out[M x 512] = A[M x 512] @ W^T (+ bias). W row-major, row stride LDW.
// f16 WMMA, f32 accumulate; one 16x16 tile per wave, 8 waves = 32M x 64N.
// ---------------------------------------------------------------------------
__global__ __launch_bounds__(256)
void proj_gemm(const float* __restrict__ A, const float* __restrict__ W,
               const float* __restrict__ bias, float* __restrict__ out) {
  const int lane = threadIdx.x & 31;
  const int wave = threadIdx.x >> 5;
  const int mt = wave >> 2, nt = wave & 3;
  const int row0 = blockIdx.x * 32 + mt * 16;
  const int col0 = blockIdx.y * 64 + nt * 16;
  const int r  = lane & 15;
  const int hi = (lane & 16) ? 1 : 0;

  const float* arow = A + (size_t)(row0 + r) * D_;
  const float* wrow = W + (size_t)(col0 + r) * LDW;

  v8f acc = {};
  for (int kk = 0; kk < D_; kk += 32) {
    const int ka = kk + hi * 8;    // A frag: hi lanes hold K=8..15,24..31
    const int kb = kk + hi * 8;    // B frag: hi lanes hold K=8..15,24..31
    v4f a0 = *(const v4f*)(arow + ka);
    v4f a1 = *(const v4f*)(arow + ka + 4);
    v4f a2 = *(const v4f*)(arow + ka + 16);
    v4f a3 = *(const v4f*)(arow + ka + 20);
    v4f b0 = *(const v4f*)(wrow + kb);
    v4f b1 = *(const v4f*)(wrow + kb + 4);
    v4f b2 = *(const v4f*)(wrow + kb + 16);
    v4f b3 = *(const v4f*)(wrow + kb + 20);
    v16h av, bv;
#pragma unroll
    for (int i = 0; i < 4; ++i) {
      av[i]      = (_Float16)a0[i];
      av[4 + i]  = (_Float16)a1[i];
      av[8 + i]  = (_Float16)a2[i];
      av[12 + i] = (_Float16)a3[i];
      bv[i]      = (_Float16)b0[i];
      bv[4 + i]  = (_Float16)b1[i];
      bv[8 + i]  = (_Float16)b2[i];
      bv[12 + i] = (_Float16)b3[i];
    }
    __builtin_prefetch(arow + kk + 32, 0, 3);   // global_prefetch_b8
    __builtin_prefetch(wrow + kk + 32, 0, 3);
    acc = __builtin_amdgcn_wmma_f32_16x16x32_f16(false, av, false, bv,
                                                 (short)0, acc, false, false);
  }
  const int c  = col0 + r;
  const float bb = bias ? bias[c] : 0.0f;
#pragma unroll
  for (int j = 0; j < 8; ++j)
    out[(size_t)(row0 + hi * 8 + j) * D_ + c] = acc[j] + bb;
}

// ---------------------------------------------------------------------------
// Partial scores: sc_ws[bq][s] = sum_h w2[h]*tanh(qp[bq][h] + kp[b][s][h])
// Block = (b, 4-q group, 256-s chunk). kp tiles (8 rows x 2KB) staged to LDS
// via TDM (one tensor_load_to_lds per tile) or per-thread async copies,
// double-buffered; each kp element reused TQ=4x from LDS.
// ---------------------------------------------------------------------------
__global__ __launch_bounds__(256)
void scores_partial(const float* __restrict__ qp, const float* __restrict__ kp,
                    const float* __restrict__ w2, float* __restrict__ sc_ws) {
  __shared__ float qpw[TQ * D_];
  __shared__ float w2s[D_];
  __shared__ float kbuf[2][TR * D_];

  const int tid  = threadIdx.x;
  const int lane = tid & 31;
  const int wave = tid >> 5;

  const int bidx = blockIdx.x;
  const int b  = bidx >> 6;
  const int qg = (bidx >> 2) & 15;
  const int sc = bidx & 3;
  const int s0 = sc * SCH;
  const int q0 = b * NQ + qg * TQ;       // first global bq row

  const float* kbase = kp + ((size_t)b * S_ + s0) * D_;

  // prologue: stage tile 0
  stage_tile(kbuf[0], kbase, tid);

  // qp rows for this q-group are contiguous (TQ*512 floats); load + w2
  const float* qbase = qp + (size_t)q0 * D_;
#pragma unroll
  for (int j = 0; j < TQ * D_ / 256; ++j) qpw[j * 256 + tid] = qbase[j * 256 + tid];
  w2s[tid]       = w2[tid];
  w2s[tid + 256] = w2[tid + 256];

  wait_stage(tid);
  __syncthreads();

  float accq[TQ];
  for (int t = 0; t < NT; ++t) {
    const int cur = t & 1;
    if (t + 1 < NT)                                   // stage tile t+1
      stage_tile(kbuf[cur ^ 1], kbase + (size_t)(t + 1) * TR * D_, tid);

    // compute tile t: wave w owns row w; lanes stride h by 32
    const float* krow = &kbuf[cur][wave * D_];
#pragma unroll
    for (int qi = 0; qi < TQ; ++qi) accq[qi] = 0.f;
#pragma unroll
    for (int h = 0; h < D_ / 32; ++h) {
      const int hh = h * 32 + lane;
      const float kv = krow[hh];
      const float wv = w2s[hh];
#pragma unroll
      for (int qi = 0; qi < TQ; ++qi)
        accq[qi] += wv * fast_tanh(qpw[qi * D_ + hh] + kv);
    }
#pragma unroll
    for (int qi = 0; qi < TQ; ++qi) {
#pragma unroll
      for (int m = 16; m > 0; m >>= 1) accq[qi] += __shfl_xor(accq[qi], m, 32);
    }
    if (lane == 0) {
      const int s = s0 + t * TR + wave;
#pragma unroll
      for (int qi = 0; qi < TQ; ++qi)
        sc_ws[(size_t)(q0 + qi) * S_ + s] = accq[qi];
    }
    wait_stage(tid);     // tile t+1 transfer complete
    __syncthreads();     // all waves done computing t; t+1 visible to all
  }
}

// ---------------------------------------------------------------------------
// Softmax over s; attn stored f16. One block per (b,q).
// ---------------------------------------------------------------------------
__global__ __launch_bounds__(256)
void softmax_kernel(const float* __restrict__ sc_ws, _Float16* __restrict__ attn) {
  __shared__ float red[256];
  const int tid = threadIdx.x;
  const int bq  = blockIdx.x;
  const float* row = sc_ws + (size_t)bq * S_;

  float v[4];
#pragma unroll
  for (int i = 0; i < 4; ++i) v[i] = row[tid + 256 * i];
  float lm = fmaxf(fmaxf(v[0], v[1]), fmaxf(v[2], v[3]));
  red[tid] = lm; __syncthreads();
  for (int off = 128; off > 0; off >>= 1) {
    if (tid < off) red[tid] = fmaxf(red[tid], red[tid + off]);
    __syncthreads();
  }
  const float mx = red[0];
  __syncthreads();

  float ls = 0.f;
#pragma unroll
  for (int i = 0; i < 4; ++i) { v[i] = __expf(v[i] - mx); ls += v[i]; }
  red[tid] = ls; __syncthreads();
  for (int off = 128; off > 0; off >>= 1) {
    if (tid < off) red[tid] += red[tid + off];
    __syncthreads();
  }
  const float inv = 1.0f / red[0];
#pragma unroll
  for (int i = 0; i < 4; ++i)
    attn[(size_t)bq * S_ + tid + 256 * i] = (_Float16)(v[i] * inv);
}

// ---------------------------------------------------------------------------
// final[b,q,v] = sum_s attn[b,q,s] * values[b,s,v]  via f16 WMMA against vT.
// ---------------------------------------------------------------------------
__global__ __launch_bounds__(256)
void av_gemm(const _Float16* __restrict__ attn, const _Float16* __restrict__ vT,
             float* __restrict__ out) {
  const int lane = threadIdx.x & 31;
  const int wave = threadIdx.x >> 5;
  const int b    = blockIdx.x >> 4;
  const int tile = (blockIdx.x & 15) * 8 + wave;  // 0..127
  const int mt = tile >> 5;                       // q tiles
  const int nt = tile & 31;                       // v tiles
  const int r  = lane & 15;
  const int hi = (lane & 16) ? 1 : 0;
  const int row0 = mt * 16;
  const int col0 = nt * 16;

  const _Float16* arow = attn + (size_t)(b * NQ + row0 + r) * S_;
  const _Float16* brow = vT + (size_t)b * D_ * S_ + (size_t)(col0 + r) * S_;

  v8f acc = {};
  for (int kk = 0; kk < S_; kk += 32) {
    const int ka = kk + hi * 16;
    const int kb = kk + hi * 16;
    v8h al = *(const v8h*)(arow + ka);
    v8h ah = *(const v8h*)(arow + ka + 8);
    v16h bv = *(const v16h*)(brow + kb);
    v16h av = __builtin_shufflevector(al, ah, 0, 1, 2, 3, 4, 5, 6, 7,
                                      8, 9, 10, 11, 12, 13, 14, 15);
    __builtin_prefetch(arow + kk + 32, 0, 3);
    __builtin_prefetch(brow + kk + 32, 0, 3);
    acc = __builtin_amdgcn_wmma_f32_16x16x32_f16(false, av, false, bv,
                                                 (short)0, acc, false, false);
  }
#pragma unroll
  for (int j = 0; j < 8; ++j)
    out[(size_t)(b * NQ + row0 + hi * 8 + j) * D_ + (col0 + r)] = acc[j];
}

// ---------------------------------------------------------------------------
extern "C" void kernel_launch(void* const* d_in, const int* in_sizes, int n_in,
                              void* d_out, int out_size, void* d_ws, size_t ws_size,
                              hipStream_t stream) {
  (void)in_sizes; (void)n_in; (void)out_size; (void)ws_size;
  const float* queries = (const float*)d_in[0];
  const float* keys    = (const float*)d_in[1];
  const float* values  = (const float*)d_in[2];
  const float* W1      = (const float*)d_in[3];
  const float* b1      = (const float*)d_in[4];
  const float* w2      = (const float*)d_in[5];
  float* out = (float*)d_out;

  char* ws = (char*)d_ws;
  float*    kp    = (float*)ws;                                   // 8 MiB
  float*    qp    = (float*)(ws + (8u << 20));                    // 512 KiB
  _Float16* attn  = (_Float16*)(ws + (8u << 20) + (512u << 10));  // 512 KiB
  _Float16* vT    = (_Float16*)(ws + (9u << 20));                 // 4 MiB
  float*    sc_ws = (float*)(ws + (13u << 20));                   // 1 MiB

  // values -> transposed f16 (B-fragment-friendly layout)
  transpose_cvt<<<dim3(S_ / 32, D_ / 32, B_), dim3(32, 8), 0, stream>>>(values, vT);
  // k_proj = keys @ W1k^T + b1   (keys is a contiguous 4096x512 matrix)
  proj_gemm<<<dim3((B_ * S_) / 32, D_ / 64), 256, 0, stream>>>(keys, W1 + 512, b1, kp);
  // q_proj = queries @ W1q^T     (contiguous 256x512)
  proj_gemm<<<dim3((B_ * NQ) / 32, D_ / 64), 256, 0, stream>>>(queries, W1, nullptr, qp);
  // partial scores with TDM/async staged kp tiles
  scores_partial<<<B_ * 64, 256, 0, stream>>>(qp, kp, w2, sc_ws);
  // softmax -> attn (f16)
  softmax_kernel<<<B_ * NQ, 256, 0, stream>>>(sc_ws, attn);
  // final = attn @ values
  av_gemm<<<B_ * 16, 256, 0, stream>>>(attn, vT, out);
}
